// MultiModalChimeraKAN_55568286875772
// MI455X (gfx1250) — matrix operand order repaired and checked by reading, not compiled, not run on hardware
//
#include <hip/hip_runtime.h>
#include <hip/hip_bf16.h>
#include <math.h>

typedef __attribute__((ext_vector_type(16))) _Float16 v16h;
typedef __attribute__((ext_vector_type(8)))  _Float16 v8h;
typedef __attribute__((ext_vector_type(8)))  float    v8f;

constexpr int kB   = 8;
constexpr int kDIM = 768;
constexpr int kDI  = 1536;
constexpr int kDS  = 16;
constexpr int kDTR = 48;
constexpr int kNG  = 8;
constexpr int kLv  = 197;
constexpr int kLa  = 32;
constexpr int kL   = 229;      // 197 + 32
constexpr int kM   = kB * kL;  // 1832
constexpr float kINVD = 1.0f / 0.33f;

__device__ __forceinline__ float dsigmoid(float x) { return 1.f / (1.f + __expf(-x)); }
__device__ __forceinline__ float dsilu(float x)    { return x * dsigmoid(x); }
__device__ __forceinline__ float dsoftplus(float x){ return (x > 20.f) ? x : log1pf(__expf(x)); }

// ---------------------------------------------------------------------------
// WMMA GEMM: C[M,N] = act( A[M,K](f16) * W[N,K](f16)^T + bias[N] )  (+= if ACCUM)
//
// Block = 256 threads (8 waves). Block tile = 16(M) x 256(N); each wave owns
// two 16x16 N-tiles and shares one A fragment across both WMMAs.
// A tile (16x32 f16, 1KB) is cooperatively staged into LDS (double-buffered,
// one s_barrier per K step) -> A global traffic /8, fragments come from
// ds_load_b128. B streams K-contiguous weight rows via global_load_b128 with
// global_prefetch_b8 ahead. K must be a multiple of 32.
//
// Fragment layouts used by this kernel (wave32; contiguous k, same for A and B):
//   A 16x32 f16 : lane L holds row M=L%16, K = 16*(L/16) + {0..15}
//   B 32x16 f16 : lane L holds col N=L%16, K = 16*(L/16) + {0..15}
//   C/D 16x16 f32: VGPR r -> row r + 8*(L/16), col L%16
// ---------------------------------------------------------------------------
template <int ACT, bool ACCUM>   // ACT: 0 none, 1 softplus
__global__ void gemm_wmma_f16(const _Float16* __restrict__ A,
                              const _Float16* __restrict__ W,
                              const float* __restrict__ bias,
                              float* __restrict__ C,
                              int M, int N, int K)
{
#if defined(__gfx1250__)
  __builtin_amdgcn_s_wait_tensorcnt(0);  // gfx1250 split-counter path (no TDM in flight)
#endif
  __shared__ _Float16 shA[2][16 * 32];

  const int lane = threadIdx.x & 31;
  const int wave = threadIdx.x >> 5;
  const int m0   = blockIdx.y * 16;
  const int n0   = (blockIdx.x * 8 + wave) * 32;   // wave covers cols [n0, n0+32)
  const int hi   = lane >> 4;   // 0 or 1
  const int lan  = lane & 15;
  const int col0 = n0 + lan;
  const int col1 = n0 + 16 + lan;
  const bool c0OK = col0 < N;
  const bool c1OK = col1 < N;

  // cooperative A staging coords: 256 threads x one u32 (2 halves) each
  const int  sr   = threadIdx.x >> 4;         // 0..15 : row within tile
  const int  sc   = (threadIdx.x & 15) * 2;   // 0..30 : even half-column
  const bool srOK = (m0 + sr) < M;
  const unsigned* aRow = (const unsigned*)(A + (size_t)(m0 + sr) * K + sc);
  const _Float16* bp0  = W + (size_t)col0 * K + hi * 16;
  const _Float16* bp1  = W + (size_t)col1 * K + hi * 16;

  v8f acc0 = {}, acc1 = {};
  int buf = 0;
  for (int k0 = 0; k0 < K; k0 += 32, buf ^= 1) {
    // stage A[m0:m0+16, k0:k0+32] -> LDS (zero-fill out-of-range rows)
    unsigned aval = srOK ? aRow[k0 >> 1] : 0u;
    *(unsigned*)&shA[buf][sr * 32 + sc] = aval;
    __syncthreads();

    // A fragment from LDS (two ds_load_b128 per lane)
    v16h a = {};
    {
      v8h alo = *(const v8h*)&shA[buf][lan * 32 + hi * 16];
      v8h ahi = *(const v8h*)&shA[buf][lan * 32 + hi * 16 + 8];
#pragma unroll
      for (int j = 0; j < 8; ++j) { a[j] = alo[j]; a[8 + j] = ahi[j]; }
    }

    // B fragments from global (weights are L2-resident at f16)
    v16h b0v = {}, b1v = {};
    if (c0OK) {
      v8h x0 = *(const v8h*)(bp0 + k0);
      v8h x1 = *(const v8h*)(bp0 + k0 + 8);
#pragma unroll
      for (int j = 0; j < 8; ++j) { b0v[j] = x0[j]; b0v[8 + j] = x1[j]; }
      __builtin_prefetch(bp0 + k0 + 64, 0, 1);   // global_prefetch_b8
    }
    if (c1OK) {
      v8h x0 = *(const v8h*)(bp1 + k0);
      v8h x1 = *(const v8h*)(bp1 + k0 + 8);
#pragma unroll
      for (int j = 0; j < 8; ++j) { b1v[j] = x0[j]; b1v[8 + j] = x1[j]; }
      __builtin_prefetch(bp1 + k0 + 64, 0, 1);
    }

    acc0 = __builtin_amdgcn_wmma_f32_16x16x32_f16(false, a, false, b0v, (short)0, acc0,
                                                  false, false);
    acc1 = __builtin_amdgcn_wmma_f32_16x16x32_f16(false, a, false, b1v, (short)0, acc1,
                                                  false, false);
  }

  const float bv0 = (c0OK && bias) ? bias[col0] : 0.f;
  const float bv1 = (c1OK && bias) ? bias[col1] : 0.f;
#pragma unroll
  for (int r = 0; r < 8; ++r) {
    const int rr = m0 + hi * 8 + r;
    if (rr >= M) continue;
    if (c0OK) {
      float v = acc0[r] + bv0;
      if (ACT == 1) v = dsoftplus(v);
      const size_t o = (size_t)rr * N + col0;
      if (ACCUM) C[o] += v; else C[o] = v;
    }
    if (c1OK) {
      float v = acc1[r] + bv1;
      if (ACT == 1) v = dsoftplus(v);
      const size_t o = (size_t)rr * N + col1;
      if (ACCUM) C[o] += v; else C[o] = v;
    }
  }
}

// ---------------------------------------------------------------------------
// Audio encoder kernels
// ---------------------------------------------------------------------------
__global__ void conv3x3_bn_relu_k(const float* __restrict__ x, const float* __restrict__ w,
                                  const float* __restrict__ cb, const float* __restrict__ bng,
                                  const float* __restrict__ bnb, float* __restrict__ y,
                                  int Ci, int Co, int H, int Wd)
{
  size_t idx = (size_t)blockIdx.x * blockDim.x + threadIdx.x;
  size_t total = (size_t)kB * Co * H * Wd;
  if (idx >= total) return;
  int ww = (int)(idx % Wd);
  int hh = (int)((idx / Wd) % H);
  int co = (int)((idx / ((size_t)Wd * H)) % Co);
  int b  = (int)(idx / ((size_t)Wd * H * Co));
  float acc = cb[co];
  for (int ci = 0; ci < Ci; ++ci) {
    const float* xp = x + ((size_t)b * Ci + ci) * H * Wd;
    const float* wp = w + ((size_t)co * Ci + ci) * 9;
#pragma unroll
    for (int ky = 0; ky < 3; ++ky) {
      int iy = hh + ky - 1;
      if ((unsigned)iy >= (unsigned)H) continue;
#pragma unroll
      for (int kx = 0; kx < 3; ++kx) {
        int ix = ww + kx - 1;
        if ((unsigned)ix >= (unsigned)Wd) continue;
        acc += xp[(size_t)iy * Wd + ix] * wp[ky * 3 + kx];
      }
    }
  }
  float s = bng[co] * rsqrtf(1.00001f);
  float v = acc * s + bnb[co];
  y[idx] = v > 0.f ? v : 0.f;
}

__global__ void maxpool2_k(const float* __restrict__ x, float* __restrict__ y,
                           int C, int H, int Wd)
{
  int Ho = H / 2, Wo = Wd / 2;
  size_t idx = (size_t)blockIdx.x * blockDim.x + threadIdx.x;
  size_t total = (size_t)kB * C * Ho * Wo;
  if (idx >= total) return;
  int wo = (int)(idx % Wo);
  int ho = (int)((idx / Wo) % Ho);
  int c  = (int)((idx / ((size_t)Wo * Ho)) % C);
  int b  = (int)(idx / ((size_t)Wo * Ho * C));
  const float* xp = x + (((size_t)b * C + c) * H + (size_t)ho * 2) * Wd + (size_t)wo * 2;
  y[idx] = fmaxf(fmaxf(xp[0], xp[1]), fmaxf(xp[Wd], xp[Wd + 1]));
}

__global__ void avgpool_k(const float* __restrict__ x, float* __restrict__ y,
                          int C, int H, int Wd, int wh, int ww)
{
  int Ho = H / wh, Wo = Wd / ww;
  size_t idx = (size_t)blockIdx.x * blockDim.x + threadIdx.x;
  size_t total = (size_t)kB * C * Ho * Wo;
  if (idx >= total) return;
  int wo = (int)(idx % Wo);
  int ho = (int)((idx / Wo) % Ho);
  int c  = (int)((idx / ((size_t)Wo * Ho)) % C);
  int b  = (int)(idx / ((size_t)Wo * Ho * C));
  const float* xp = x + (((size_t)b * C + c) * H + (size_t)ho * wh) * Wd + (size_t)wo * ww;
  float s = 0.f;
  for (int i = 0; i < wh; ++i)
    for (int j = 0; j < ww; ++j) s += xp[(size_t)i * Wd + j];
  y[idx] = s / (float)(wh * ww);
}

// (B,256,4,8) -> A[(b*32+hw)][c] f16, K=256
__global__ void pool_to_a16_k(const float* __restrict__ p, _Float16* __restrict__ A)
{
  int idx = blockIdx.x * blockDim.x + threadIdx.x;
  if (idx >= kB * 256 * 32) return;
  int c  = idx % 256;
  int hw = (idx / 256) % 32;
  int b  = idx / (256 * 32);
  A[((size_t)(b * 32 + hw)) * 256 + c] = (_Float16)p[((size_t)(b * 256 + c)) * 32 + hw];
}

// ---------------------------------------------------------------------------
// LayerNorm (D=768): one block per row, LDS tree reductions; optional f16 copy
// and optional per-row mean-of-output ("energy").
// ---------------------------------------------------------------------------
__global__ void layernorm_k(const float* __restrict__ x, const float* __restrict__ g,
                            const float* __restrict__ b, float* __restrict__ y,
                            _Float16* __restrict__ y16, float* __restrict__ energy, int M)
{
  __shared__ float s0[256];
  __shared__ float s1[256];
  const int rowI = blockIdx.x;
  if (rowI >= M) return;
  const float* xr = x + (size_t)rowI * kDIM;
  const int t = threadIdx.x;
  float sum = 0.f, sq = 0.f;
  for (int i = t; i < kDIM; i += 256) { float v = xr[i]; sum += v; sq += v * v; }
  s0[t] = sum; s1[t] = sq; __syncthreads();
  for (int off = 128; off > 0; off >>= 1) {
    if (t < off) { s0[t] += s0[t + off]; s1[t] += s1[t + off]; }
    __syncthreads();
  }
  const float mean = s0[0] / (float)kDIM;
  const float var  = s1[0] / (float)kDIM - mean * mean;
  const float inv  = rsqrtf(var + 1e-5f);
  __syncthreads();
  float esum = 0.f;
  for (int i = t; i < kDIM; i += 256) {
    float v = (xr[i] - mean) * inv * g[i] + b[i];
    if (y)   y[(size_t)rowI * kDIM + i] = v;
    if (y16) y16[(size_t)rowI * kDIM + i] = (_Float16)v;
    esum += v;
  }
  if (energy) {
    s0[t] = esum; __syncthreads();
    for (int off = 128; off > 0; off >>= 1) {
      if (t < off) s0[t] += s0[t + off];
      __syncthreads();
    }
    if (t == 0) energy[rowI] = s0[0] / (float)kDIM;
  }
}

// aud_global[b,d] = mean_t aud_ln[(b*32+t), d]
__global__ void aud_global_k(const float* __restrict__ aud_ln, float* __restrict__ out)
{
  int idx = blockIdx.x * blockDim.x + threadIdx.x;
  if (idx >= kB * kDIM) return;
  int b = idx / kDIM, d = idx % kDIM;
  float s = 0.f;
  for (int tk = 0; tk < kLa; ++tk) s += aud_ln[((size_t)(b * kLa + tk)) * kDIM + d];
  out[idx] = s / (float)kLa;
}

// FKAN basis: z[m,d] -> basis[m, d*8+g] = 1 - tanh^2((z - grid_g)/0.33), f16
__global__ void fkan_basis_k(const float* __restrict__ z, _Float16* __restrict__ basis, int M)
{
  size_t idx = (size_t)blockIdx.x * blockDim.x + threadIdx.x;
  size_t total = (size_t)M * kDIM;
  if (idx >= total) return;
  float zv = z[idx];
#pragma unroll
  for (int g = 0; g < kNG; ++g) {
    float gr = -2.f + (float)g * (4.f / 7.f);
    float tt = tanhf((zv - gr) * kINVD);
    basis[idx * kNG + g] = (_Float16)(1.f - tt * tt);
  }
}

__global__ void f32_to_f16_k(const float* __restrict__ s, _Float16* __restrict__ d, size_t n)
{
  size_t i = (size_t)blockIdx.x * blockDim.x + threadIdx.x;
  if (i < n) d[i] = (_Float16)s[i];
}

// dt_w (1536 x 48) -> f16 (1536 x 64) zero-padded so GEMM K%32==0
__global__ void pad_dtw_k(const float* __restrict__ s, _Float16* __restrict__ d)
{
  int idx = blockIdx.x * blockDim.x + threadIdx.x;
  if (idx >= kDI * 64) return;
  int r = idx / 64, c = idx % 64;
  d[idx] = (c < kDTR) ? (_Float16)s[r * kDTR + c] : (_Float16)0.f;
}

// weight[b] = sigmoid(aud_global[b] . gw + gb)
__global__ void gate_k(const float* __restrict__ ag, const float* __restrict__ gw,
                       const float* __restrict__ gb, float* __restrict__ gate)
{
  int b = threadIdx.x;
  if (b >= kB) return;
  float acc = gb[0];
  for (int j = 0; j < kDIM; ++j) acc += ag[b * kDIM + j] * gw[j];
  gate[b] = dsigmoid(acc);
}

// full_energy: linear interp of aud_energy (8 x 32) to (8 x 229)
__global__ void interp_energy_k(const float* __restrict__ ae, float* __restrict__ fe)
{
  int idx = blockIdx.x * blockDim.x + threadIdx.x;
  if (idx >= kB * kL) return;
  int b = idx / kL, l = idx % kL;
  float pos = ((float)l + 0.5f) * ((float)kLa / (float)kL) - 0.5f;
  pos = fminf(fmaxf(pos, 0.f), (float)(kLa - 1));
  int lo = (int)floorf(pos);
  int hi = min(lo + 1, kLa - 1);
  float w = pos - (float)lo;
  fe[idx] = ae[b * kLa + lo] * (1.f - w) + ae[b * kLa + hi] * w;
}

// comb = [pixel + gate*ghost + tok_vis ; aud_ln + tok_aud]
__global__ void assemble_comb_k(const float* __restrict__ pixel, const float* __restrict__ aud_ln,
                                const float* __restrict__ ghost, const float* __restrict__ gate,
                                const float* __restrict__ tokv, const float* __restrict__ toka,
                                float* __restrict__ comb)
{
  size_t idx = (size_t)blockIdx.x * blockDim.x + threadIdx.x;
  if (idx >= (size_t)kM * kDIM) return;
  int d = (int)(idx % kDIM);
  int l = (int)((idx / kDIM) % kL);
  int b = (int)(idx / ((size_t)kDIM * kL));
  float v;
  if (l < kLv)
    v = pixel[((size_t)(b * kLv + l)) * kDIM + d] + gate[b] * ghost[b * kDIM + d] + tokv[d];
  else
    v = aud_ln[((size_t)(b * kLa + (l - kLv))) * kDIM + d] + toka[d];
  comb[idx] = v;
}

// depthwise causal conv1d (k=4, left pad 3) + SiLU; input = first half of xz
__global__ void dwconv_silu_k(const float* __restrict__ xz, const float* __restrict__ cw,
                              const float* __restrict__ cbi, float* __restrict__ xc,
                              _Float16* __restrict__ xc16)
{
  size_t idx = (size_t)blockIdx.x * blockDim.x + threadIdx.x;
  if (idx >= (size_t)kM * kDI) return;
  int d = (int)(idx % kDI);
  int l = (int)((idx / kDI) % kL);
  int b = (int)(idx / ((size_t)kDI * kL));
  float acc = cbi[d];
#pragma unroll
  for (int t = 0; t < 4; ++t) {
    int ll = l - 3 + t;
    if (ll >= 0) acc += cw[d * 4 + t] * xz[((size_t)(b * kL + ll)) * (2 * kDI) + d];
  }
  float v = dsilu(acc);
  xc[idx] = v;
  xc16[idx] = (_Float16)v;
}

// delta_in (padded to K=64): j<48 -> dbc[:,j] + (1/(fe+1e-4))*e2w[j] + e2b[j]; else 0
__global__ void delta_in_k(const float* __restrict__ dbc, const float* __restrict__ fe,
                           const float* __restrict__ e2w, const float* __restrict__ e2b,
                           _Float16* __restrict__ din)
{
  size_t idx = (size_t)blockIdx.x * blockDim.x + threadIdx.x;
  if (idx >= (size_t)kM * 64) return;
  int j = (int)(idx % 64);
  size_t r = idx / 64;
  float v = 0.f;
  if (j < kDTR) {
    float inv = 1.f / (fe[r] + 1e-4f);
    v = dbc[r * 80 + j] + inv * e2w[j] + e2b[j];
  }
  din[idx] = (_Float16)v;
}

// selective-scan: per (b,d), 16 states in registers; emits gated output in f16
__global__ void scan_k(const float* __restrict__ delta, const float* __restrict__ dbc,
                       const float* __restrict__ xc, const float* __restrict__ xz,
                       const float* __restrict__ Alog, const float* __restrict__ Dp,
                       _Float16* __restrict__ yg)
{
  int idx = blockIdx.x * blockDim.x + threadIdx.x;
  if (idx >= kB * kDI) return;
  int d = idx % kDI, b = idx / kDI;
  float As[kDS], h[kDS];
#pragma unroll
  for (int s = 0; s < kDS; ++s) { As[s] = -__expf(Alog[d * kDS + s]); h[s] = 0.f; }
  const float Dd = Dp[d];
  for (int l = 0; l < kL; ++l) {
    size_t r = (size_t)(b * kL + l);
    float dl  = delta[r * kDI + d];
    float xcv = xc[r * kDI + d];
    float y = 0.f;
#pragma unroll
    for (int s = 0; s < kDS; ++s) {
      float da = __expf(dl * As[s]);
      h[s] = da * h[s] + dl * dbc[r * 80 + kDTR + s] * xcv;
      y += h[s] * dbc[r * 80 + kDTR + kDS + s];
    }
    float zv = xz[r * (2 * kDI) + kDI + d];
    yg[r * kDI + d] = (_Float16)((y + Dd * xcv) * dsilu(zv));
  }
}

// pooled = mean over L; write f32 + f16 + directly into d_out segment
__global__ void pooled_k(const float* __restrict__ comb, float* __restrict__ pooledf,
                         _Float16* __restrict__ pooled16, float* __restrict__ dout)
{
  int idx = blockIdx.x * blockDim.x + threadIdx.x;
  if (idx >= kB * kDIM) return;
  int b = idx / kDIM, d = idx % kDIM;
  float s = 0.f;
  for (int l = 0; l < kL; ++l) s += comb[((size_t)(b * kL + l)) * kDIM + d];
  float m = s / (float)kL;
  pooledf[idx] = m;
  pooled16[idx] = (_Float16)m;
  dout[872 + idx] = m;   // pooled segment: after emo(56)+ind(800)+gen(16)
}

// ---------------------------------------------------------------------------
// Host side
// ---------------------------------------------------------------------------
static inline int cdiv(long long a, long long b) { return (int)((a + b - 1) / b); }

static void gemm(const _Float16* A, const _Float16* W, const float* bias, float* C,
                 int M, int N, int K, int act, bool accum, hipStream_t st)
{
  dim3 blk(256, 1, 1);
  dim3 grd(cdiv(N, 256), cdiv(M, 16), 1);   // block tile: 16 x 256
  if (accum)         gemm_wmma_f16<0, true ><<<grd, blk, 0, st>>>(A, W, bias, C, M, N, K);
  else if (act == 1) gemm_wmma_f16<1, false><<<grd, blk, 0, st>>>(A, W, bias, C, M, N, K);
  else               gemm_wmma_f16<0, false><<<grd, blk, 0, st>>>(A, W, bias, C, M, N, K);
}

static void cvt(const float* s, _Float16* d, size_t n, hipStream_t st)
{
  f32_to_f16_k<<<cdiv((long long)n, 256), 256, 0, st>>>(s, d, n);
}

extern "C" void kernel_launch(void* const* d_in, const int* in_sizes, int n_in,
                              void* d_out, int out_size, void* d_ws, size_t ws_size,
                              hipStream_t stream)
{
  (void)in_sizes; (void)n_in; (void)out_size; (void)ws_size;
  auto P = [&](int i) { return (const float*)d_in[i]; };

  const float* pixel = P(0);
  const float* audio = P(1);
  // JAX pytree flatten order: dict keys sorted alphabetically, lists in order.
  // ae: 2..21
  const float *bn1b=P(2),*bn1g=P(3),*bn2b=P(4),*bn2g=P(5),*bn3b=P(6),*bn3g=P(7),
              *bn4b=P(8),*bn4g=P(9),*c1b=P(10),*c1w=P(11),*c2b=P(12),*c2w=P(13),
              *c3b=P(14),*c3w=P(15),*c4b=P(16),*c4w=P(17),*ae_lnb=P(18),*ae_lng=P(19),
              *pb=P(20),*pw=P(21);
  const float *ce_lnb=P(22),*ce_lng=P(23),*ce_sw=P(24);           // cls_emo
  const float *cg_lnb=P(25),*cg_lng=P(26),*cg_sw=P(27);           // cls_gen
  const float *ci_lnb=P(28),*ci_lng=P(29),*ci_sw=P(30);           // cls_ind
  const float *ik_gb=P(31),*ik_gw=P(32),*ik_lnb=P(33),*ik_lng=P(34),*ik_sw=P(35);
  const float *proj_be_b=P(80),*proj_be_w=P(81),*proj_id_b=P(82),*proj_id_w=P(83);
  const float *tok_aud=P(84),*tok_vis=P(85);

  // ---- workspace bump allocator -------------------------------------------
  char* base = (char*)d_ws;
  size_t off = 0;
  auto allocf = [&](size_t n) { float* r = (float*)(base + off);
    off = (off + n * sizeof(float) + 255) & ~(size_t)255; return r; };
  auto alloch = [&](size_t n) { _Float16* r = (_Float16*)(base + off);
    off = (off + n * sizeof(_Float16) + 255) & ~(size_t)255; return r; };

  float* conv_a    = allocf(16777216);       // ping (largest conv activation)
  float* conv_b    = allocf(4194304);        // pong
  float* p4        = allocf(65536);
  float* aud_proj  = allocf((size_t)256 * kDIM);
  float* aud_ln    = allocf((size_t)256 * kDIM);
  float* aud_energy= allocf(256);
  float* aud_glob  = allocf(kB * kDIM);
  float* z8        = allocf(kB * kDIM);
  float* ghost     = allocf(kB * kDIM);
  float* gateb     = allocf(kB);
  float* comb      = allocf((size_t)kM * kDIM);
  float* fe        = allocf(kM);
  float* lnA       = allocf((size_t)kM * kDIM);
  float* lnB       = allocf((size_t)kM * kDIM);
  float* xz        = allocf((size_t)kM * 2 * kDI);
  float* xc        = allocf((size_t)kM * kDI);
  float* dbc       = allocf((size_t)kM * 80);
  float* delta     = allocf((size_t)kM * kDI);
  float* pooledf   = allocf(kB * kDIM);
  float* feat_id   = allocf(kB * kDIM);
  float* feat_emo  = allocf(kB * kDIM);
  _Float16* audA16   = alloch((size_t)256 * 256);
  _Float16* basis8   = alloch((size_t)kB * kDIM * kNG);
  _Float16* ln16     = alloch((size_t)kM * kDIM);
  _Float16* xc16     = alloch((size_t)kM * kDI);
  _Float16* din16    = alloch((size_t)kM * 64);
  _Float16* yg16     = alloch((size_t)kM * kDI);
  _Float16* basis16  = alloch((size_t)kM * kDIM * kNG);
  _Float16* wbuf16   = alloch((size_t)kDIM * kDIM * kNG);  // staged f16 weights (reused)
  _Float16* pooled16 = alloch(kB * kDIM);

  float* outp = (float*)d_out;

  // ---- audio encoder ------------------------------------------------------
  conv3x3_bn_relu_k<<<cdiv(16777216, 256), 256, 0, stream>>>(audio, c1w, c1b, bn1g, bn1b,
                                                             conv_a, 1, 32, 128, 512);
  maxpool2_k<<<cdiv(4194304, 256), 256, 0, stream>>>(conv_a, conv_b, 32, 128, 512);
  conv3x3_bn_relu_k<<<cdiv(8388608, 256), 256, 0, stream>>>(conv_b, c2w, c2b, bn2g, bn2b,
                                                            conv_a, 32, 64, 64, 256);
  maxpool2_k<<<cdiv(2097152, 256), 256, 0, stream>>>(conv_a, conv_b, 64, 64, 256);
  conv3x3_bn_relu_k<<<cdiv(4194304, 256), 256, 0, stream>>>(conv_b, c3w, c3b, bn3g, bn3b,
                                                            conv_a, 64, 128, 32, 128);
  maxpool2_k<<<cdiv(1048576, 256), 256, 0, stream>>>(conv_a, conv_b, 128, 32, 128);
  conv3x3_bn_relu_k<<<cdiv(2097152, 256), 256, 0, stream>>>(conv_b, c4w, c4b, bn4g, bn4b,
                                                            conv_a, 128, 256, 16, 64);
  avgpool_k<<<cdiv(65536, 256), 256, 0, stream>>>(conv_a, p4, 256, 16, 64, 4, 8);
  pool_to_a16_k<<<cdiv(65536, 256), 256, 0, stream>>>(p4, audA16);

  cvt(pw, wbuf16, (size_t)kDIM * 256, stream);
  gemm(audA16, wbuf16, pb, aud_proj, 256, kDIM, 256, 0, false, stream);
  layernorm_k<<<256, 256, 0, stream>>>(aud_proj, ae_lng, ae_lnb, aud_ln, nullptr,
                                       aud_energy, 256);
  aud_global_k<<<cdiv(kB * kDIM, 256), 256, 0, stream>>>(aud_ln, aud_glob);

  // ikan ghost = fkan(aud_global) ; weight = sigmoid(aud_global @ gw.T + gb)
  layernorm_k<<<kB, 256, 0, stream>>>(aud_glob, ik_lng, ik_lnb, z8, nullptr, nullptr, kB);
  fkan_basis_k<<<cdiv(kB * kDIM, 256), 256, 0, stream>>>(z8, basis8, kB);
  cvt(ik_sw, wbuf16, (size_t)kDIM * kDIM * kNG, stream);
  gemm(basis8, wbuf16, nullptr, ghost, kB, kDIM, kDIM * kNG, 0, false, stream);
  gate_k<<<1, 32, 0, stream>>>(aud_glob, ik_gw, ik_gb, gateb);

  assemble_comb_k<<<cdiv((long long)kM * kDIM, 256), 256, 0, stream>>>(
      pixel, aud_ln, ghost, gateb, tok_vis, tok_aud, comb);
  interp_energy_k<<<cdiv(kB * kL, 256), 256, 0, stream>>>(aud_energy, fe);

  // ---- 2x (Mamba + FKAN) blocks ------------------------------------------
  for (int layer = 0; layer < 2; ++layer) {
    const int Lb = 36 + layer * 22;   // sorted: fkan{lnb,lng,sw}, mamba{...}, n1b..n2g
    const float *fk_lnb=P(Lb+0),*fk_lng=P(Lb+1),*fk_sw=P(Lb+2);
    const float *A_log=P(Lb+3),*Dp=P(Lb+4),*conv_bi=P(Lb+5),*conv_w=P(Lb+6),
                *dt_b=P(Lb+7),*dt_w=P(Lb+8),*e2dt_b=P(Lb+9),*e2dt_w=P(Lb+10),
                *in_b=P(Lb+11),*in_w=P(Lb+12),*m_nb=P(Lb+13),*m_ng=P(Lb+14),
                *out_b=P(Lb+15),*out_w=P(Lb+16),*xp_w=P(Lb+17);
    const float *n1b=P(Lb+18),*n1g=P(Lb+19),*n2b=P(Lb+20),*n2g=P(Lb+21);

    // mamba( layernorm(comb, n1) ): inner LN(ng,nb) -> f16 activations
    layernorm_k<<<kM, 256, 0, stream>>>(comb, n1g, n1b, lnA, nullptr, nullptr, kM);
    layernorm_k<<<kM, 256, 0, stream>>>(lnA, m_ng, m_nb, nullptr, ln16, nullptr, kM);

    cvt(in_w, wbuf16, (size_t)2 * kDI * kDIM, stream);
    gemm(ln16, wbuf16, in_b, xz, kM, 2 * kDI, kDIM, 0, false, stream);

    dwconv_silu_k<<<cdiv((long long)kM * kDI, 256), 256, 0, stream>>>(xz, conv_w, conv_bi,
                                                                      xc, xc16);
    cvt(xp_w, wbuf16, (size_t)(kDTR + 2 * kDS) * kDI, stream);
    gemm(xc16, wbuf16, nullptr, dbc, kM, kDTR + 2 * kDS, kDI, 0, false, stream);

    delta_in_k<<<cdiv((long long)kM * 64, 256), 256, 0, stream>>>(dbc, fe, e2dt_w, e2dt_b,
                                                                  din16);
    pad_dtw_k<<<cdiv(kDI * 64, 256), 256, 0, stream>>>(dt_w, wbuf16);
    gemm(din16, wbuf16, dt_b, delta, kM, kDI, 64, /*softplus*/1, false, stream);

    scan_k<<<cdiv(kB * kDI, 256), 256, 0, stream>>>(delta, dbc, xc, xz, A_log, Dp, yg16);

    cvt(out_w, wbuf16, (size_t)kDIM * kDI, stream);
    gemm(yg16, wbuf16, out_b, comb, kM, kDIM, kDI, 0, /*residual accumulate*/true, stream);

    // comb += fkan( layernorm(comb, n2), fk_lng/lnb, fk_sw )
    layernorm_k<<<kM, 256, 0, stream>>>(comb, n2g, n2b, lnA, nullptr, nullptr, kM);
    layernorm_k<<<kM, 256, 0, stream>>>(lnA, fk_lng, fk_lnb, lnB, nullptr, nullptr, kM);
    fkan_basis_k<<<cdiv((long long)kM * kDIM, 256), 256, 0, stream>>>(lnB, basis16, kM);
    cvt(fk_sw, wbuf16, (size_t)kDIM * kDIM * kNG, stream);
    gemm(basis16, wbuf16, nullptr, comb, kM, kDIM, kDIM * kNG, 0, true, stream);
  }

  // ---- heads --------------------------------------------------------------
  pooled_k<<<cdiv(kB * kDIM, 256), 256, 0, stream>>>(comb, pooledf, pooled16, outp);

  cvt(proj_id_w, wbuf16, (size_t)kDIM * kDIM, stream);
  gemm(pooled16, wbuf16, proj_id_b, feat_id, kB, kDIM, kDIM, 0, false, stream);
  cvt(proj_be_w, wbuf16, (size_t)kDIM * kDIM, stream);
  gemm(pooled16, wbuf16, proj_be_b, feat_emo, kB, kDIM, kDIM, 0, false, stream);

  // emo = fkan(feat_emo, cls_emo) -> d_out[0 : 56)
  layernorm_k<<<kB, 256, 0, stream>>>(feat_emo, ce_lng, ce_lnb, z8, nullptr, nullptr, kB);
  fkan_basis_k<<<cdiv(kB * kDIM, 256), 256, 0, stream>>>(z8, basis8, kB);
  cvt(ce_sw, wbuf16, (size_t)7 * kDIM * kNG, stream);
  gemm(basis8, wbuf16, nullptr, outp + 0, kB, 7, kDIM * kNG, 0, false, stream);

  // ind = fkan(feat_id, cls_ind) -> d_out[56 : 856)
  layernorm_k<<<kB, 256, 0, stream>>>(feat_id, ci_lng, ci_lnb, z8, nullptr, nullptr, kB);
  fkan_basis_k<<<cdiv(kB * kDIM, 256), 256, 0, stream>>>(z8, basis8, kB);
  cvt(ci_sw, wbuf16, (size_t)100 * kDIM * kNG, stream);
  gemm(basis8, wbuf16, nullptr, outp + 56, kB, 100, kDIM * kNG, 0, false, stream);

  // gen = fkan(feat_id, cls_gen) -> d_out[856 : 872)
  layernorm_k<<<kB, 256, 0, stream>>>(feat_id, cg_lng, cg_lnb, z8, nullptr, nullptr, kB);
  fkan_basis_k<<<cdiv(kB * kDIM, 256), 256, 0, stream>>>(z8, basis8, kB);
  cvt(cg_sw, wbuf16, (size_t)2 * kDIM * kNG, stream);
  gemm(basis8, wbuf16, nullptr, outp + 856, kB, 2, kDIM * kNG, 0, false, stream);
}
